// LocAtt_20486994002113
// MI455X (gfx1250) — matrix operand order repaired and checked by reading, not compiled, not run on hardware
//
#include <hip/hip_runtime.h>
#include <hip/hip_bf16.h>

// ---------------------------------------------------------------------------
// Flash-attention for MI455X (gfx1250, wave32, WMMA).
// B=2 H=32 Q=2048 D=128, fp32 in/out, bf16 matrix-core math, fp32 accum/softmax.
// Grid: (Q/BR, B*H) blocks of 256 threads (8 waves); each wave owns 16 rows.
// Round 2: batched B-fragment loads (4 ds-load pairs ahead of 4 WMMAs) to
// break the per-WMMA s_wait_dscnt chain, kc-outer loops so consecutive WMMAs
// share the A operand, and global_prefetch of the next KV tiles.
// ---------------------------------------------------------------------------

typedef __attribute__((ext_vector_type(16))) __bf16 v16bf;
typedef __attribute__((ext_vector_type(8)))  float  v8f;

#define B_   2
#define H_   32
#define Q_   2048
#define D_   128
#define BR   128          // query rows per block (8 waves x 16 rows)
#define BC   128          // kv columns per iteration
#define PITCH 136         // LDS row pitch in bf16 elements (128 + 8 pad -> 272B, staggers banks)
#define SCALE 0.08838834764831845f     // 1/sqrt(128)
#define NEG_HUGE (-3.4028234663852886e38f)  // finfo(float32).min

union FragA {             // one wave-striped 16x32 (A) or 32x16 (B) bf16 matrix
    v16bf bf;
    uint4 q[2];
};

#define ZERO8 ((v8f){0.f,0.f,0.f,0.f,0.f,0.f,0.f,0.f})

__global__ __launch_bounds__(256) void flash_attn_bf16_wmma(
    const float* __restrict__ q_in,       // [B, Q, H, D]
    const float* __restrict__ k_in,       // [B, Q, H, D]
    const float* __restrict__ v_in,       // [B, Q, H, D]
    const float* __restrict__ alibi,      // [B*H, 1, Q]
    const unsigned char* __restrict__ mask, // [B, 1, Q, Q] bool
    float* __restrict__ out)              // [B, Q, H*D]
{
    extern __shared__ __align__(16) unsigned char smem_raw[];
    __bf16* sK = (__bf16*)smem_raw;        // [BC][PITCH]  K rows (kv, d)
    __bf16* sV = sK + BC * PITCH;          // [D ][PITCH]  V transposed (d, kv)
    __bf16* sP = sV + D_ * PITCH;          // [BR][PITCH]  probs, per-wave-private rows

    const int mt   = blockIdx.x;           // row tile
    const int bh   = blockIdx.y;           // head index (b*H + h)
    const int bb   = bh / H_;
    const int hh   = bh % H_;
    const int tid  = threadIdx.x;
    const int wave = tid >> 5;
    const int lane = tid & 31;
    const int ln16 = lane & 15;
    const int hl   = lane >> 4;            // which 16-lane half

    const int q0 = mt * BR;
    const int m0 = wave * 16;              // wave's row offset inside tile

    // ---- Q A-fragments: fp32 global -> bf16, ISA A-layout, loaded once ----
    // lane<16: M=ln16, K {0..7}U{16..23}; lane>=16: M=ln16, K {8..15}U{24..31}
    FragA aQ[4];
    {
        const int qrow = q0 + m0 + ln16;
        const float* qp = q_in + (((size_t)bb * Q_ + qrow) * H_ + hh) * D_;
        #pragma unroll
        for (int kc = 0; kc < 4; ++kc) {
            const int k0 = kc * 32 + hl * 8;
            float4 f0 = *(const float4*)(qp + k0);
            float4 f1 = *(const float4*)(qp + k0 + 4);
            float4 f2 = *(const float4*)(qp + k0 + 16);
            float4 f3 = *(const float4*)(qp + k0 + 20);
            aQ[kc].bf[0]  = (__bf16)f0.x; aQ[kc].bf[1]  = (__bf16)f0.y;
            aQ[kc].bf[2]  = (__bf16)f0.z; aQ[kc].bf[3]  = (__bf16)f0.w;
            aQ[kc].bf[4]  = (__bf16)f1.x; aQ[kc].bf[5]  = (__bf16)f1.y;
            aQ[kc].bf[6]  = (__bf16)f1.z; aQ[kc].bf[7]  = (__bf16)f1.w;
            aQ[kc].bf[8]  = (__bf16)f2.x; aQ[kc].bf[9]  = (__bf16)f2.y;
            aQ[kc].bf[10] = (__bf16)f2.z; aQ[kc].bf[11] = (__bf16)f2.w;
            aQ[kc].bf[12] = (__bf16)f3.x; aQ[kc].bf[13] = (__bf16)f3.y;
            aQ[kc].bf[14] = (__bf16)f3.z; aQ[kc].bf[15] = (__bf16)f3.w;
        }
    }

    // mask row base pointers for this wave's 16 rows (C-layout: vgpr r + half*8)
    const unsigned char* maskRow[8];
    #pragma unroll
    for (int r = 0; r < 8; ++r)
        maskRow[r] = mask + ((size_t)bb * Q_ + (q0 + m0 + hl * 8 + r)) * Q_;

    // online-softmax state per C-layout row, replicated across each 16-lane half
    float mR[8], lR[8];
    v8f   accO[8];
    #pragma unroll
    for (int r = 0; r < 8; ++r) { mR[r] = NEG_HUGE; lR[r] = 0.f; }
    #pragma unroll
    for (int dt = 0; dt < 8; ++dt) accO[dt] = ZERO8;

    for (int kv0 = 0; kv0 < Q_; kv0 += BC) {
        // ---- prefetch next iteration's K/V tiles toward L2/L0 while we compute
        if (kv0 + BC < Q_) {
            const int pr = tid >> 1;             // 2 threads per kv row
            const int po = (tid & 1) * 64;       // 64 floats = 256 B apart
            const float* kp = k_in + (((size_t)bb * Q_ + kv0 + BC + pr) * H_ + hh) * D_ + po;
            const float* vp = v_in + (((size_t)bb * Q_ + kv0 + BC + pr) * H_ + hh) * D_ + po;
            __builtin_prefetch(kp, 0, 1);
            __builtin_prefetch(kp + 32, 0, 1);   // next 128B line
            __builtin_prefetch(vp, 0, 1);
            __builtin_prefetch(vp + 32, 0, 1);
        }

        // ---- cooperative stage: K tile (row-major) fp32 -> bf16 LDS ----
        for (int i = tid; i < BC * (D_ / 4); i += 256) {
            const int kv = i >> 5;
            const int dv = (i & 31) * 4;
            float4 f = *(const float4*)(k_in + (((size_t)bb * Q_ + kv0 + kv) * H_ + hh) * D_ + dv);
            union { __bf16 e[4]; uint2 u; } pk;
            pk.e[0] = (__bf16)f.x; pk.e[1] = (__bf16)f.y;
            pk.e[2] = (__bf16)f.z; pk.e[3] = (__bf16)f.w;
            *(uint2*)&sK[kv * PITCH + dv] = pk.u;
        }
        // ---- cooperative stage: V tile transposed (d, kv) fp32 -> bf16 LDS ----
        for (int i = tid; i < BC * (D_ / 4); i += 256) {
            const int kv = i >> 5;
            const int dv = (i & 31) * 4;
            float4 f = *(const float4*)(v_in + (((size_t)bb * Q_ + kv0 + kv) * H_ + hh) * D_ + dv);
            sV[(dv + 0) * PITCH + kv] = (__bf16)f.x;
            sV[(dv + 1) * PITCH + kv] = (__bf16)f.y;
            sV[(dv + 2) * PITCH + kv] = (__bf16)f.z;
            sV[(dv + 3) * PITCH + kv] = (__bf16)f.w;
        }
        __syncthreads();

        // ---- S = Q * K^T  (16 rows x 128 cols, fp32 accum) ----
        // kc outer: consecutive WMMAs share aQ[kc]; B-fragments loaded in
        // groups of 4 so 8 ds_load_b128 are in flight ahead of 4 WMMAs.
        v8f accS[8];
        #pragma unroll
        for (int nt = 0; nt < 8; ++nt) accS[nt] = ZERO8;
        #pragma unroll
        for (int kc = 0; kc < 4; ++kc) {
            #pragma unroll
            for (int g = 0; g < 2; ++g) {
                FragA bK[4];
                #pragma unroll
                for (int j = 0; j < 4; ++j) {
                    const __bf16* kb = sK + ((g * 4 + j) * 16 + ln16) * PITCH + kc * 32 + hl * 8;
                    bK[j].q[0] = *(const uint4*)(kb);
                    bK[j].q[1] = *(const uint4*)(kb + 16);
                }
                #pragma unroll
                for (int j = 0; j < 4; ++j)
                    accS[g * 4 + j] = __builtin_amdgcn_wmma_f32_16x16x32_bf16(
                        false, aQ[kc].bf, false, bK[j].bf, (short)0, accS[g * 4 + j], false, false);
            }
        }

        // ---- bias + mask ----
        #pragma unroll
        for (int nt = 0; nt < 8; ++nt) {
            const int colg = kv0 + nt * 16 + ln16;
            const float al = alibi[(size_t)bh * Q_ + colg];
            #pragma unroll
            for (int r = 0; r < 8; ++r) {
                float s = al + SCALE * accS[nt][r];
                if (maskRow[r][colg]) s = NEG_HUGE;
                accS[nt][r] = s;
            }
        }

        // ---- online softmax per row (16-lane butterfly inside each half) ----
        #pragma unroll
        for (int r = 0; r < 8; ++r) {
            float rm = accS[0][r];
            #pragma unroll
            for (int nt = 1; nt < 8; ++nt) rm = fmaxf(rm, accS[nt][r]);
            #pragma unroll
            for (int off = 1; off < 16; off <<= 1)
                rm = fmaxf(rm, __shfl_xor(rm, off, 16));
            const float mnew = fmaxf(mR[r], rm);
            const float cf   = __expf(mR[r] - mnew);
            mR[r] = mnew;
            #pragma unroll
            for (int dt = 0; dt < 8; ++dt) accO[dt][r] *= cf;
            float rs = 0.f;
            #pragma unroll
            for (int nt = 0; nt < 8; ++nt) {
                float p = __expf(accS[nt][r] - mnew);
                accS[nt][r] = p;
                rs += p;
            }
            #pragma unroll
            for (int off = 1; off < 16; off <<= 1)
                rs += __shfl_xor(rs, off, 16);
            lR[r] = lR[r] * cf + rs;
        }

        // ---- P (C-layout f32) -> LDS bf16 (wave-private rows, no barrier) ----
        #pragma unroll
        for (int nt = 0; nt < 8; ++nt)
            #pragma unroll
            for (int r = 0; r < 8; ++r)
                sP[(m0 + hl * 8 + r) * PITCH + nt * 16 + ln16] = (__bf16)accS[nt][r];

        // ---- O += P * V  (same batched-fragment structure) ----
        #pragma unroll
        for (int kc = 0; kc < 4; ++kc) {
            FragA aP;
            const __bf16* pb = sP + (m0 + ln16) * PITCH + kc * 32 + hl * 8;
            aP.q[0] = *(const uint4*)(pb);
            aP.q[1] = *(const uint4*)(pb + 16);
            #pragma unroll
            for (int g = 0; g < 2; ++g) {
                FragA bV[4];
                #pragma unroll
                for (int j = 0; j < 4; ++j) {
                    const __bf16* vb = sV + ((g * 4 + j) * 16 + ln16) * PITCH + kc * 32 + hl * 8;
                    bV[j].q[0] = *(const uint4*)(vb);
                    bV[j].q[1] = *(const uint4*)(vb + 16);
                }
                #pragma unroll
                for (int j = 0; j < 4; ++j)
                    accO[g * 4 + j] = __builtin_amdgcn_wmma_f32_16x16x32_bf16(
                        false, aP.bf, false, bV[j].bf, (short)0, accO[g * 4 + j], false, false);
            }
        }
        __syncthreads();   // protect sK/sV before next iteration's stores
    }

    // ---- epilogue: O / l, merge heads -> out[b, q, h*D + d] ----
    #pragma unroll
    for (int r = 0; r < 8; ++r) {
        const float inv  = 1.0f / lR[r];
        const int   rowg = q0 + m0 + hl * 8 + r;
        float* op = out + ((size_t)bb * Q_ + rowg) * (H_ * D_) + hh * D_ + ln16;
        #pragma unroll
        for (int dt = 0; dt < 8; ++dt)
            op[dt * 16] = accO[dt][r] * inv;
    }
}

extern "C" void kernel_launch(void* const* d_in, const int* in_sizes, int n_in,
                              void* d_out, int out_size, void* d_ws, size_t ws_size,
                              hipStream_t stream) {
    (void)in_sizes; (void)n_in; (void)d_ws; (void)ws_size; (void)out_size;
    const float*         q_in  = (const float*)d_in[0];
    const float*         k_in  = (const float*)d_in[1];
    const float*         v_in  = (const float*)d_in[2];
    const float*         alibi = (const float*)d_in[3];
    const unsigned char* mask  = (const unsigned char*)d_in[4];
    // d_in[5] (num_heads) / d_in[6] (head_dim) are compile-time constants here.
    float* out = (float*)d_out;

    dim3 grid(Q_ / BR, B_ * H_);
    const size_t shmem = (size_t)(BC + D_ + BR) * PITCH * sizeof(__bf16);  // 102 KB
    hipLaunchKernelGGL(flash_attn_bf16_wmma, grid, dim3(256), shmem, stream,
                       q_in, k_in, v_in, alibi, mask, out);
}
